// GNNStream_19748259627008
// MI455X (gfx1250) — matrix operand order repaired and checked by reading, not compiled, not run on hardware
//
#include <hip/hip_runtime.h>
#include <hip/hip_bf16.h>
#include <math.h>

typedef __attribute__((ext_vector_type(16))) _Float16 v16h;
typedef __attribute__((ext_vector_type(8)))  float    v8f;

#define B_    8
#define S_    2048
#define CLIP_ 512
#define H_    512
#define BN_   256
#define NC_   10

// gfx1250 async global->LDS path (CDNA5 ISA) -- accepted
// by the assembler in round 2.
#define USE_ASYNC_STAGE 1

__device__ __forceinline__ void async_copy16(unsigned lds_off, const void* gptr) {
#if USE_ASYNC_STAGE
  asm volatile("global_load_async_to_lds_b128 %0, %1, off"
               :: "v"(lds_off), "v"(gptr) : "memory");
#endif
}
__device__ __forceinline__ void async_wait0() {
#if USE_ASYNC_STAGE
  asm volatile("s_wait_asynccnt 0x0" ::: "memory");
#endif
}

// ---------------------------------------------------------------------------
// WMMA helpers (wave32, 16x16x32 f16 -> f32). Layouts used here (A/B k order is this file's own choice):
//  A (16x32, MxK): lane m(0..15)/half: elems i -> K = k0+16*half+i  (contiguous)
//  B (32x16, KxN): lane n = lane&15, elems i -> K = k0+16*half+i  (contiguous)
//  C/D (16x16):    lane n = lane&15, elem v -> M = v + 8*half
// ---------------------------------------------------------------------------
__device__ __forceinline__ v16h load_a_frag(const _Float16* A, int lda, int k0, int lane) {
  const int m = lane & 15, half = lane >> 4;
  const _Float16* p0 = A + (size_t)m * lda + k0 + 16 * half;
  const _Float16* p1 = p0 + 8;
  v16h a;
#pragma unroll
  for (int i = 0; i < 8; ++i) { a[i] = p0[i]; a[i + 8] = p1[i]; }
  return a;
}

__device__ __forceinline__ v16h load_b_frag(const _Float16* WT, int ldb, int k0, int lane) {
  const int n = lane & 15, half = lane >> 4;
  const _Float16* p = WT + (size_t)n * ldb + k0 + 16 * half;
  v16h b;
#pragma unroll
  for (int i = 0; i < 16; ++i) b[i] = p[i];
  return b;
}

__device__ __forceinline__ v8f wmma16(v16h a, v16h b, v8f c) {
  return __builtin_amdgcn_wmma_f32_16x16x32_f16(false, a, false, b, (short)0, c, false, false);
}

// ---------------------------------------------------------------------------
// Weight convert+transpose: w[K][N] f32 -> wT[N][K] f16
// ---------------------------------------------------------------------------
__global__ void k_wt(const float* __restrict__ w, _Float16* __restrict__ wT, int K, int N) {
  int i = blockIdx.x * 256 + threadIdx.x;
  if (i < K * N) {
    int kk = i / N, n = i - kk * N;
    wT[(size_t)n * K + kk] = (_Float16)w[i];
  }
}

// ---------------------------------------------------------------------------
// Encoder: h = relu(LN(x @ w_enc + b_enc)) -> f16 [B*S][H]
// 128 threads (4 waves) per 16-row tile; A fragments register-cached.
// launch_bounds(.,1): one block per WGP by design -> full VGPR budget,
// keeps all 16 A fragments resident (no scratch spills).
// ---------------------------------------------------------------------------
__global__ void __launch_bounds__(128, 1)
k_encode(const float* __restrict__ x, const _Float16* __restrict__ wT,
         const float* __restrict__ bias, const float* __restrict__ g,
         const float* __restrict__ beta, _Float16* __restrict__ hout) {
  extern __shared__ char smem[];
  _Float16* xt   = (_Float16*)smem;                              // 16x512 f16
  float*    accb = (float*)(smem + 16 * CLIP_ * 2);              // 16x512 f32
  float*    st   = (float*)(smem + 16 * CLIP_ * 2 + 16 * H_ * 4);// stats+partials
  float*    ps   = st + 32;        // [16][8] partial sums
  float*    ps2  = ps + 128;       // [16][8] partial sumsq
  const int tid = threadIdx.x, lane = tid & 31, wave = tid >> 5;
  const size_t r0 = (size_t)blockIdx.x * 16;

  for (int i = tid; i < 16 * CLIP_; i += 128)
    xt[i] = (_Float16)x[r0 * CLIP_ + i];          // tile rows are contiguous
  __syncthreads();

  const int n0 = lane & 15, half = lane >> 4;

  // register-cache the loop-invariant A fragments (16 x 8 VGPRs = 128 VGPRs)
  v16h afr[CLIP_ / 32];
#pragma unroll
  for (int ks = 0; ks < CLIP_ / 32; ++ks) afr[ks] = load_a_frag(xt, CLIP_, ks * 32, lane);

  for (int nt = wave; nt < H_ / 16; nt += 4) {
    v8f acc = {};
#pragma unroll
    for (int ks = 0; ks < CLIP_ / 32; ++ks)
      acc = wmma16(afr[ks], load_b_frag(wT + (size_t)nt * 16 * CLIP_, CLIP_, ks * 32, lane), acc);
    int n = nt * 16 + n0;
    float bn = bias[n];
#pragma unroll
    for (int v = 0; v < 8; ++v) accb[(v + 8 * half) * H_ + n] = acc[v] + bn;
  }
  __syncthreads();

  {  // LN stats: 16 rows x 8 chunks of 64
    int row = tid >> 3, chunk = tid & 7;
    float s = 0.f, s2 = 0.f;
    for (int j = chunk * 64; j < chunk * 64 + 64; ++j) {
      float v = accb[row * H_ + j];
      s += v; s2 += v * v;
    }
    ps[row * 8 + chunk] = s; ps2[row * 8 + chunk] = s2;
  }
  __syncthreads();
  if (tid < 16) {
    float s = 0.f, s2 = 0.f;
    for (int c = 0; c < 8; ++c) { s += ps[tid * 8 + c]; s2 += ps2[tid * 8 + c]; }
    float mu = s * (1.0f / H_);
    float var = s2 * (1.0f / H_) - mu * mu;
    st[tid] = mu;
    st[16 + tid] = rsqrtf(var + 1e-5f);
  }
  __syncthreads();
  for (int i = tid; i < 16 * H_; i += 128) {
    int m = i >> 9, n = i & (H_ - 1);
    float val = (accb[i] - st[m]) * st[16 + m] * g[n] + beta[n];
    hout[r0 * H_ + i] = (_Float16)fmaxf(val, 0.f);
  }
}

// ---------------------------------------------------------------------------
// q/k/v projections: one block per 16-row tile, h tile staged once in LDS
// (async copy), A fragments register-cached across all 24 jobs per wave.
// v stored transposed vT[b][n][s] for contiguous attn@v B-fragments.
// ---------------------------------------------------------------------------
__global__ void __launch_bounds__(128, 1)
k_qkv(const _Float16* __restrict__ h, const _Float16* __restrict__ wTq,
      const _Float16* __restrict__ wTk, const _Float16* __restrict__ wTv,
      const float* __restrict__ bq, const float* __restrict__ bk,
      const float* __restrict__ bv, _Float16* __restrict__ qo,
      _Float16* __restrict__ ko, _Float16* __restrict__ vT) {
  extern __shared__ char smem[];
  _Float16* ht = (_Float16*)smem;  // 16x512 f16 = 16KB at LDS offset 0
  const int tid = threadIdx.x, lane = tid & 31, wave = tid >> 5;
  const size_t r0 = (size_t)blockIdx.x * 16;
  const int bidx = (int)(r0 / S_);
  const int srow = (int)(r0 % S_);

#if USE_ASYNC_STAGE
  for (int c = tid; c < 16 * H_ * 2 / 16; c += 128)
    async_copy16((unsigned)(c * 16), (const char*)(h + r0 * H_) + c * 16);
  async_wait0();
#else
  for (int i = tid; i < 16 * H_; i += 128) ht[i] = h[r0 * H_ + i];
#endif
  __syncthreads();

  const int n0 = lane & 15, half = lane >> 4;

  v16h afr[H_ / 32];
#pragma unroll
  for (int ks = 0; ks < H_ / 32; ++ks) afr[ks] = load_a_frag(ht, H_, ks * 32, lane);

  for (int job = wave; job < 96; job += 4) {
    const int which = job >> 5, nt = job & 31;
    const _Float16* wT = which == 0 ? wTq : (which == 1 ? wTk : wTv);
    const float* bias  = which == 0 ? bq  : (which == 1 ? bk  : bv);
    v8f acc = {};
#pragma unroll
    for (int ks = 0; ks < H_ / 32; ++ks)
      acc = wmma16(afr[ks], load_b_frag(wT + (size_t)nt * 16 * H_, H_, ks * 32, lane), acc);
    int n = nt * 16 + n0;
    float bn = bias[n];
#pragma unroll
    for (int v = 0; v < 8; ++v) {
      int m = v + 8 * half;
      float val = acc[v] + bn;
      if (which == 0)      qo[(r0 + m) * H_ + n] = (_Float16)val;
      else if (which == 1) ko[(r0 + m) * H_ + n] = (_Float16)val;
      else                 vT[((size_t)bidx * H_ + n) * S_ + (srow + m)] = (_Float16)val;
    }
  }
}

// ---------------------------------------------------------------------------
// Fused attention per (batch, 16-query tile): 4 waves, ~226KB of the WGP's
// 320KB LDS. scores strip -> softmax -> spatial gate + mask -> attn@v ->
// bottleneck GEMM + LN + relu -> out f32 [B*S][BN].
// Q-tile A-fragments register-cached across all 32 key tiles per wave.
// ---------------------------------------------------------------------------
__global__ void __launch_bounds__(128, 1)
k_attn(const _Float16* __restrict__ qb, const _Float16* __restrict__ kb,
       const _Float16* __restrict__ vT, const float* __restrict__ coords,
       const unsigned char* __restrict__ mask,
       const _Float16* __restrict__ wT_bn, const float* __restrict__ b_bn,
       const float* __restrict__ g2, const float* __restrict__ beta2,
       float* __restrict__ outb) {
  extern __shared__ char smem[];
  float*    scores = (float*)smem;                                             // 128 KB
  _Float16* attnh  = (_Float16*)(smem + 16 * S_ * 4);                          //  64 KB
  _Float16* tileA  = (_Float16*)(smem + 16 * S_ * 4 + 16 * S_ * 2);            //  16 KB
  float*    outt   = (float*)(smem + 16 * S_ * 4 + 16 * S_ * 2 + 16 * H_ * 2); //  16 KB
  float*    st     = (float*)(smem + 16 * S_ * 4 + 16 * S_ * 2 + 16 * H_ * 2 + 16 * BN_ * 4);
  float*    pmax   = st + 32;    // [16][8]
  float*    psum   = pmax + 128; // [16][8]

  const int tid = threadIdx.x, lane = tid & 31, wave = tid >> 5;
  const int n0 = lane & 15, half = lane >> 4;
  const size_t r0 = (size_t)blockIdx.x * 16;
  const int bidx = (int)(r0 / S_);
  const int s0 = (int)(r0 % S_);

  // stage Q tile into LDS (f16, contiguous 16KB) -- async global->LDS
  const unsigned tA_off = (unsigned)(16 * S_ * 4 + 16 * S_ * 2);
#if USE_ASYNC_STAGE
  for (int c = tid; c < 16 * H_ * 2 / 16; c += 128)
    async_copy16(tA_off + c * 16, (const char*)(qb + r0 * H_) + c * 16);
  async_wait0();
#else
  for (int i = tid; i < 16 * H_; i += 128) tileA[i] = qb[r0 * H_ + i];
#endif
  __syncthreads();

  // register-cache Q-tile A fragments (16 x 8 VGPRs)
  {
    v16h qfr[H_ / 32];
#pragma unroll
    for (int ks = 0; ks < H_ / 32; ++ks) qfr[ks] = load_a_frag(tileA, H_, ks * 32, lane);

    // scores = q @ k^T / sqrt(H): waves split the 128 key tiles
    const float inv = 1.0f / sqrtf((float)H_);
    for (int t = wave; t < S_ / 16; t += 4) {
      v8f acc = {};
      const _Float16* kbase = kb + ((size_t)bidx * S_ + t * 16) * H_;
      __builtin_prefetch(kbase + 4 * 16 * H_, 0, 0);
#pragma unroll
      for (int ks = 0; ks < H_ / 32; ++ks)
        acc = wmma16(qfr[ks], load_b_frag(kbase, H_, ks * 32, lane), acc);
#pragma unroll
      for (int v = 0; v < 8; ++v) scores[(v + 8 * half) * S_ + t * 16 + n0] = acc[v] * inv;
    }
  }
  __syncthreads();

  // softmax row stats, 8-way parallel per row (softmax over ALL columns)
  {
    int row = tid >> 3, chunk = tid & 7;
    float mx = -1e30f;
    for (int t = chunk * (S_ / 8); t < (chunk + 1) * (S_ / 8); ++t)
      mx = fmaxf(mx, scores[row * S_ + t]);
    pmax[row * 8 + chunk] = mx;
  }
  __syncthreads();
  if (tid < 16) {
    float mx = -1e30f;
    for (int c = 0; c < 8; ++c) mx = fmaxf(mx, pmax[tid * 8 + c]);
    st[tid] = mx;
  }
  __syncthreads();
  {
    int row = tid >> 3, chunk = tid & 7;
    float s = 0.f, mx = st[row];
    for (int t = chunk * (S_ / 8); t < (chunk + 1) * (S_ / 8); ++t)
      s += expf(scores[row * S_ + t] - mx);
    psum[row * 8 + chunk] = s;
  }
  __syncthreads();
  if (tid < 16) {
    float s = 0.f;
    for (int c = 0; c < 8; ++c) s += psum[tid * 8 + c];
    st[16 + tid] = s;
  }
  __syncthreads();

  // attn = softmax * exp(-5*dist) with masked columns zeroed -> f16
  for (int i = tid; i < 16 * S_; i += 128) {
    int m = i >> 11, t = i & (S_ - 1);
    float p = expf(scores[i] - st[m]) / st[16 + m];
    float qx = coords[((size_t)bidx * S_ + s0 + m) * 3 + 0];
    float qy = coords[((size_t)bidx * S_ + s0 + m) * 3 + 1];
    float kx = coords[((size_t)bidx * S_ + t) * 3 + 0];
    float ky = coords[((size_t)bidx * S_ + t) * 3 + 1];
    float dx = qx - kx, dy = qy - ky;
    float sq = dx * dx + dy * dy;
    float dist = sq > 0.f ? sqrtf(sq) : 0.f;
    float adj = expf(-5.0f * dist);
    attnh[i] = (_Float16)(mask[bidx * S_ + t] ? 0.f : p * adj);
  }
  __syncthreads();

  // h_graph = attn @ v (K=2048: stream A from LDS; waves split 32 N-tiles)
  for (int nt = wave; nt < H_ / 16; nt += 4) {
    v8f acc = {};
    const _Float16* vbase = vT + ((size_t)bidx * H_ + nt * 16) * S_;
#pragma unroll 2
    for (int k0 = 0; k0 < S_; k0 += 32) {
      v16h a = load_a_frag(attnh, S_, k0, lane);
      v16h b = load_b_frag(vbase, S_, k0, lane);
      acc = wmma16(a, b, acc);
    }
    int n = nt * 16 + n0;
#pragma unroll
    for (int v = 0; v < 8; ++v) tileA[(v + 8 * half) * H_ + n] = (_Float16)acc[v];
  }
  __syncthreads();

  // pre-LN bottleneck: h_graph @ w_bn + b_bn (A fragments register-cached)
  {
    v16h hfr[H_ / 32];
#pragma unroll
    for (int ks = 0; ks < H_ / 32; ++ks) hfr[ks] = load_a_frag(tileA, H_, ks * 32, lane);

    for (int nt = wave; nt < BN_ / 16; nt += 4) {
      v8f acc = {};
#pragma unroll
      for (int ks = 0; ks < H_ / 32; ++ks)
        acc = wmma16(hfr[ks], load_b_frag(wT_bn + (size_t)nt * 16 * H_, H_, ks * 32, lane), acc);
      int n = nt * 16 + n0;
      float bn = b_bn[n];
#pragma unroll
      for (int v = 0; v < 8; ++v) outt[(v + 8 * half) * BN_ + n] = acc[v] + bn;
    }
  }
  __syncthreads();

  if (tid < 16) {  // LN2 stats over 256 (short; keep serial per row)
    float s = 0.f, s2 = 0.f;
    for (int j = 0; j < BN_; ++j) { float v = outt[tid * BN_ + j]; s += v; s2 += v * v; }
    float mu = s * (1.0f / BN_);
    float var = s2 * (1.0f / BN_) - mu * mu;
    st[32 + tid] = mu;
    st[48 + tid] = rsqrtf(var + 1e-5f);
  }
  __syncthreads();
  for (int i = tid; i < 16 * BN_; i += 128) {
    int m = i >> 8, n = i & (BN_ - 1);
    float val = (outt[i] - st[32 + m]) * st[48 + m] * g2[n] + beta2[n];
    outb[((size_t)bidx * S_ + s0) * BN_ + i] = fmaxf(val, 0.f);
  }
}

// ---------------------------------------------------------------------------
// Masked mean pooling: pooled[b][n]
// ---------------------------------------------------------------------------
__global__ void k_pool(const float* __restrict__ outb, const unsigned char* __restrict__ mask,
                       float* __restrict__ pooled) {
  int b = blockIdx.x, n = threadIdx.x;
  float sum = 0.f, cnt = 0.f;
  for (int s = 0; s < S_; ++s) {
    if (!mask[b * S_ + s]) { sum += outb[((size_t)b * S_ + s) * BN_ + n]; cnt += 1.f; }
  }
  pooled[b * BN_ + n] = sum / fmaxf(cnt, 1e-9f);
}

// ---------------------------------------------------------------------------
// Classifier head (tiny): logits = relu(pooled @ w_c1 + b_c1) @ w_c2 + b_c2
// ---------------------------------------------------------------------------
__global__ void k_head(const float* __restrict__ pooled, const float* __restrict__ w1,
                       const float* __restrict__ b1, const float* __restrict__ w2,
                       const float* __restrict__ b2, float* __restrict__ logits) {
  int idx = threadIdx.x;
  if (idx >= B_ * NC_) return;
  int b = idx / NC_, c = idx - b * NC_;
  float acc = b2[c];
  for (int j = 0; j < BN_ / 2; ++j) {
    float hj = b1[j];
    for (int n = 0; n < BN_; ++n) hj += pooled[b * BN_ + n] * w1[n * (BN_ / 2) + j];
    acc += fmaxf(hj, 0.f) * w2[j * NC_ + c];
  }
  logits[idx] = acc;
}

// ---------------------------------------------------------------------------
extern "C" void kernel_launch(void* const* d_in, const int* in_sizes, int n_in,
                              void* d_out, int out_size, void* d_ws, size_t ws_size,
                              hipStream_t stream) {
  const float* local_feat = (const float*)d_in[0];
  const float* coords     = (const float*)d_in[1];
  const unsigned char* mask = (const unsigned char*)d_in[2];
  const float* w_enc = (const float*)d_in[3];
  const float* b_enc = (const float*)d_in[4];
  const float* ln1_g = (const float*)d_in[5];
  const float* ln1_b = (const float*)d_in[6];
  const float* wq = (const float*)d_in[7];
  const float* bq = (const float*)d_in[8];
  const float* wk = (const float*)d_in[9];
  const float* bk = (const float*)d_in[10];
  const float* wv = (const float*)d_in[11];
  const float* bv = (const float*)d_in[12];
  const float* w_bn = (const float*)d_in[13];
  const float* b_bn = (const float*)d_in[14];
  const float* ln2_g = (const float*)d_in[15];
  const float* ln2_b = (const float*)d_in[16];
  const float* w_c1 = (const float*)d_in[17];
  const float* b_c1 = (const float*)d_in[18];
  const float* w_c2 = (const float*)d_in[19];
  const float* b_c2 = (const float*)d_in[20];

  char* w = (char*)d_ws;
  _Float16* wT_enc = (_Float16*)w; w += (size_t)H_ * CLIP_ * 2;
  _Float16* wT_q   = (_Float16*)w; w += (size_t)H_ * H_ * 2;
  _Float16* wT_k   = (_Float16*)w; w += (size_t)H_ * H_ * 2;
  _Float16* wT_v   = (_Float16*)w; w += (size_t)H_ * H_ * 2;
  _Float16* wT_bn  = (_Float16*)w; w += (size_t)BN_ * H_ * 2;
  _Float16* hbuf   = (_Float16*)w; w += (size_t)B_ * S_ * H_ * 2;
  _Float16* qbuf   = (_Float16*)w; w += (size_t)B_ * S_ * H_ * 2;
  _Float16* kbuf   = (_Float16*)w; w += (size_t)B_ * S_ * H_ * 2;
  _Float16* vTbuf  = (_Float16*)w; w += (size_t)B_ * S_ * H_ * 2;
  float*    outbuf = (float*)w;    w += (size_t)B_ * S_ * BN_ * 4;

  float* logits = (float*)d_out;
  float* pooled = (float*)d_out + B_ * NC_;

  // weight conversion / transposition to f16
  k_wt<<<(CLIP_ * H_ + 255) / 256, 256, 0, stream>>>(w_enc, wT_enc, CLIP_, H_);
  k_wt<<<(H_ * H_ + 255) / 256, 256, 0, stream>>>(wq, wT_q, H_, H_);
  k_wt<<<(H_ * H_ + 255) / 256, 256, 0, stream>>>(wk, wT_k, H_, H_);
  k_wt<<<(H_ * H_ + 255) / 256, 256, 0, stream>>>(wv, wT_v, H_, H_);
  k_wt<<<(H_ * BN_ + 255) / 256, 256, 0, stream>>>(w_bn, wT_bn, H_, BN_);

  // encoder (4 waves per tile)
  size_t enc_lds = (size_t)16 * CLIP_ * 2 + (size_t)16 * H_ * 4 + 2048;
  k_encode<<<B_ * S_ / 16, 128, enc_lds, stream>>>(local_feat, wT_enc, b_enc, ln1_g, ln1_b, hbuf);

  // q / k / v (one block per tile; 4 waves split 96 jobs; h staged once)
  k_qkv<<<B_ * S_ / 16, 128, (size_t)16 * H_ * 2, stream>>>(hbuf, wT_q, wT_k, wT_v,
                                                            bq, bk, bv, qbuf, kbuf, vTbuf);

  // fused attention + bottleneck (~226KB of the WGP's 320KB LDS, 4 waves)
  size_t attn_lds = (size_t)16 * S_ * 4 + (size_t)16 * S_ * 2 + (size_t)16 * H_ * 2 +
                    (size_t)16 * BN_ * 4 + 2048;
  (void)hipFuncSetAttribute((const void*)k_attn, hipFuncAttributeMaxDynamicSharedMemorySize,
                            (int)attn_lds);
  k_attn<<<B_ * S_ / 16, 128, attn_lds, stream>>>(qbuf, kbuf, vTbuf, coords, mask, wT_bn, b_bn,
                                                  ln2_g, ln2_b, outbuf);

  // pooling + head
  k_pool<<<B_, BN_, 0, stream>>>(outbuf, mask, pooled);
  k_head<<<1, 128, 0, stream>>>(pooled, w_c1, b_c1, w_c2, b_c2, logits);
}
